// GATFraudDetector_34660386078848
// MI455X (gfx1250) — matrix operand order repaired and checked by reading, not compiled, not run on hardware
//
#include <hip/hip_runtime.h>
#include <hip/hip_bf16.h>

typedef __attribute__((ext_vector_type(16))) _Float16 v16h;
typedef __attribute__((ext_vector_type(8)))  float    v8f;

#if __has_builtin(__builtin_amdgcn_tensor_load_to_lds) && \
    __has_builtin(__builtin_amdgcn_s_wait_tensorcnt)
#define USE_TDM 1
typedef unsigned int t_v4u __attribute__((ext_vector_type(4)));
typedef int          t_v8i __attribute__((ext_vector_type(8)));
typedef int          t_v4i __attribute__((ext_vector_type(4)));
#endif

// ---------------------------------------------------------------------------
// Generic fill
// ---------------------------------------------------------------------------
__global__ void k_fill(float* p, long long n, float v) {
    long long i = (long long)blockIdx.x * blockDim.x + threadIdx.x;
    if (i < n) p[i] = v;
}

// ---------------------------------------------------------------------------
// WMMA GEMM: C[M x N] = A[M x K] @ B[K x N], fp32 in memory, f16 multiply,
// fp32 accumulate. Block = 8 waves covering a 128-row x 16-col tile; the
// K x 16 B panel is staged once per block into LDS (via the Tensor Data
// Mover when available) and shared by all 8 waves.
// ---------------------------------------------------------------------------
__global__ void __launch_bounds__(256)
k_gemm_wmma(const float* __restrict__ A,
            const float* __restrict__ B,
            float* __restrict__ C,
            int M, int N, int K) {
    __shared__ float Bs[256 * 16];          // K (<=256) x 16 f32 panel, 16 KB

    int tid   = threadIdx.x;
    int ntile = N >> 4;
    int bn    = blockIdx.x % ntile;         // 16-col tile
    int bmblk = blockIdx.x / ntile;         // 128-row block
    int col0  = bn * 16;

#ifdef USE_TDM
    if ((tid >> 5) == 0) {                  // wave 0 issues the tensor DMA
        unsigned lds_off = (unsigned)(uintptr_t)&Bs[0];     // low 32 bits = LDS byte addr
        unsigned long long ga = (unsigned long long)(uintptr_t)(B + col0);
        t_v4u g0;
        g0[0] = 1u;                                          // count=1 (valid D#)
        g0[1] = lds_off;                                     // lds_addr
        g0[2] = (unsigned)(ga & 0xFFFFFFFFu);                // global_addr[31:0]
        g0[3] = (unsigned)((ga >> 32) & 0x01FFFFFFu)         // global_addr[56:32]
                | 0x80000000u;                               // type=2 ("image")
        t_v8i g1;
        g1[0] = 0x20000;                                     // data_size=2 -> 4 bytes
        g1[1] = (int)(((unsigned)N & 0xFFFFu) << 16);        // tensor_dim0 lo16 @bits[63:48]
        g1[2] = (int)((((unsigned)N >> 16) & 0xFFFFu)        // tensor_dim0 hi16
                | (((unsigned)K & 0xFFFFu) << 16));          // tensor_dim1 lo16
        g1[3] = (int)((((unsigned)K >> 16) & 0xFFFFu)        // tensor_dim1 hi16
                | (16u << 16));                              // tile_dim0 = 16
        g1[4] = (int)((unsigned)K & 0xFFFFu);                // tile_dim1 = K, tile_dim2 = 0
        g1[5] = N;                                           // tensor_dim0_stride[31:0]
        g1[6] = 0;                                           // stride hi / dim1_stride lo
        g1[7] = 0;
        t_v4i z4 = {};
#if defined(__clang_major__) && __clang_major__ >= 23
        t_v8i z8 = {};
        __builtin_amdgcn_tensor_load_to_lds(g0, g1, z4, z4, z8, 0);
#else
        __builtin_amdgcn_tensor_load_to_lds(g0, g1, z4, z4, 0);
#endif
        __builtin_amdgcn_s_wait_tensorcnt(0);
    }
#else
    for (int i = tid; i < K * 16; i += 256) {
        int k = i >> 4, n = i & 15;
        Bs[i] = B[(size_t)k * N + col0 + n];
    }
#endif
    __syncthreads();

    int wv   = tid >> 5;
    int lane = tid & 31;
    int tm   = bmblk * 8 + wv;              // this wave's 16-row tile
    if (tm * 16 >= M) return;               // wave-uniform exit (EXEC stays full)

    int l15 = lane & 15;
    int hi  = lane >> 4;                    // lane group 0 / 1
    int row = tm * 16 + l15;                // A row this lane feeds

    v8f acc = {};
#pragma unroll 2
    for (int k0 = 0; k0 < K; k0 += 32) {
        // ---- A fragment (16x32 f16, contiguous k order: element s holds k = 16*hi + s) ----
        const float* arow = A + (size_t)row * K + k0;
        __builtin_prefetch(arow + 32, 0, 0);   // global_prefetch_b8 next K chunk
        int ab = hi ? 16 : 0;                  // lanes 16-31 start at K=16
        v16h a;
#pragma unroll
        for (int v = 0; v < 4; ++v) {
            a[2*v]         = (_Float16)arow[ab + 2*v];
            a[2*v + 1]     = (_Float16)arow[ab + 2*v + 1];
            a[8 + 2*v]     = (_Float16)arow[8 + ab + 2*v];
            a[8 + 2*v + 1] = (_Float16)arow[8 + ab + 2*v + 1];
        }
        // ---- B fragment from the LDS panel: lane = column, K halves by group ----
        const float* bp = &Bs[(k0 + hi * 16) * 16 + l15];
        v16h b;
#pragma unroll
        for (int v = 0; v < 8; ++v) {
            b[2*v]     = (_Float16)bp[(2*v)     * 16];
            b[2*v + 1] = (_Float16)bp[(2*v + 1) * 16];
        }
        acc = __builtin_amdgcn_wmma_f32_16x16x32_f16(
            /*neg_a=*/false, a, /*neg_b=*/false, b,
            /*c_mod=*/(short)0, acc, /*reuse_a=*/false, /*reuse_b=*/false);
    }
    // ---- C/D layout: VGPR v -> row (v + 8*hi), column = col0 + l15 ----
    float* crow = C + (size_t)(tm * 16 + hi * 8) * N + col0 + l15;
#pragma unroll
    for (int v = 0; v < 8; ++v) crow[(size_t)v * N] = acc[v];
}

// ---------------------------------------------------------------------------
// Per-(node,head) attention logits: als = sum_c h*a_s, ald = sum_c h*a_d
// ---------------------------------------------------------------------------
__global__ void k_logits(const float* __restrict__ h,
                         const float* __restrict__ a_s,
                         const float* __restrict__ a_d,
                         float* __restrict__ als, float* __restrict__ ald,
                         int Nn, int H, int C) {
    int idx = blockIdx.x * blockDim.x + threadIdx.x;   // node*H + head
    if (idx >= Nn * H) return;
    int hh = idx % H;
    const float* hr = h + (size_t)(idx / H) * H * C + (size_t)hh * C;
    float s = 0.f, d = 0.f;
    for (int c = 0; c < C; ++c) {
        float v = hr[c];
        s += v * a_s[hh * C + c];
        d += v * a_d[hh * C + c];
    }
    als[idx] = s;
    ald[idx] = d;
}

// ---------------------------------------------------------------------------
// float atomic max via signed/unsigned int punning
// ---------------------------------------------------------------------------
__device__ inline void atomicMaxF(float* a, float v) {
    if (v >= 0.f) atomicMax((int*)a, __float_as_int(v));
    else          atomicMin((unsigned int*)a, __float_as_uint(v));
}

__device__ inline float leaky02(float v) { return v >= 0.f ? v : 0.2f * v; }

// Pass 1: segment max of leaky-relu logits over dst (incl. self loops)
__global__ void k_edge_max(const int* __restrict__ ei, long long E, int Nn,
                           const float* __restrict__ als,
                           const float* __restrict__ ald,
                           float* __restrict__ m, int H) {
    long long i = (long long)blockIdx.x * blockDim.x + threadIdx.x;
    long long tot = (E + Nn) * H;
    if (i >= tot) return;
    int hh = (int)(i % H);
    long long w = i / H;
    int src, dst;
    if (w < E) { src = ei[w]; dst = ei[E + w]; } else { src = dst = (int)(w - E); }
    float e = leaky02(als[(long long)src * H + hh] + ald[(long long)dst * H + hh]);
    atomicMaxF(&m[(long long)dst * H + hh], e);
}

// Pass 2: denom = segment sum of exp(e - m[dst])
__global__ void k_edge_denom(const int* __restrict__ ei, long long E, int Nn,
                             const float* __restrict__ als,
                             const float* __restrict__ ald,
                             const float* __restrict__ m,
                             float* __restrict__ den, int H) {
    long long i = (long long)blockIdx.x * blockDim.x + threadIdx.x;
    long long tot = (E + Nn) * H;
    if (i >= tot) return;
    int hh = (int)(i % H);
    long long w = i / H;
    int src, dst;
    if (w < E) { src = ei[w]; dst = ei[E + w]; } else { src = dst = (int)(w - E); }
    float e = leaky02(als[(long long)src * H + hh] + ald[(long long)dst * H + hh]);
    atomicAdd(&den[(long long)dst * H + hh], __expf(e - m[(long long)dst * H + hh]));
}

// Pass 3: out[dst] += alpha * h[src]; one wave32 per edge, lanes cover 4
// contiguous channels each -> global_load_b128 reads + f32 global atomics.
__global__ void k_edge_scatter(const int* __restrict__ ei, long long E, int Nn,
                               const float* __restrict__ als,
                               const float* __restrict__ ald,
                               const float* __restrict__ m,
                               const float* __restrict__ den,
                               const float* __restrict__ h,
                               float* __restrict__ out, int H, int HC) {
    long long w = (long long)blockIdx.x * (blockDim.x >> 5) + (threadIdx.x >> 5);
    long long Etot = E + Nn;
    if (w >= Etot) return;
    int lane = threadIdx.x & 31;
    int src, dst;
    if (w < E) { src = ei[w]; dst = ei[E + w]; } else { src = dst = (int)(w - E); }

    float alpha[4];
    for (int hh = 0; hh < H; ++hh) {
        float e = leaky02(als[(long long)src * H + hh] + ald[(long long)dst * H + hh]);
        alpha[hh] = __expf(e - m[(long long)dst * H + hh]) /
                    (den[(long long)dst * H + hh] + 1e-16f);
    }
    int C = HC / H;
    const float4* hs4 = (const float4*)(h + (size_t)src * HC);
    float* od = out + (size_t)dst * HC;
    for (int c4 = lane; c4 * 4 < HC; c4 += 32) {
        float4 v = hs4[c4];
        float  al = alpha[(c4 * 4) / C];     // 4 | C, so one head per float4
        int    c  = c4 * 4;
        atomicAdd(&od[c + 0], v.x * al);
        atomicAdd(&od[c + 1], v.y * al);
        atomicAdd(&od[c + 2], v.z * al);
        atomicAdd(&od[c + 3], v.w * al);
    }
}

// ---------------------------------------------------------------------------
// BN statistics: per-channel sum / sumsq via LDS float atomics (ds_add_f32),
// one global atomicAdd per (block, channel).
// ---------------------------------------------------------------------------
__global__ void k_bn_stats(const float* __restrict__ x,
                           float* __restrict__ sums, float* __restrict__ sqs,
                           int Nn, int HC) {
    __shared__ float ls[256];
    __shared__ float lq[256];
    int t = threadIdx.x;
    if (t < HC) { ls[t] = 0.f; lq[t] = 0.f; }
    __syncthreads();
    long long total  = (long long)Nn * HC;
    long long stride = (long long)gridDim.x * blockDim.x;
    for (long long i = (long long)blockIdx.x * blockDim.x + t; i < total; i += stride) {
        float v = x[i];
        int c = (int)(i % HC);
        atomicAdd(&ls[c], v);
        atomicAdd(&lq[c], v * v);
    }
    __syncthreads();
    if (t < HC) {
        atomicAdd(&sums[t], ls[t]);
        atomicAdd(&sqs[t],  lq[t]);
    }
}

// BN normalize + ELU, in place (per-channel GAT bias cancels under BN)
__global__ void k_bn_apply(float* __restrict__ x,
                           const float* __restrict__ sums,
                           const float* __restrict__ sqs,
                           const float* __restrict__ g,
                           const float* __restrict__ be,
                           int Nn, int HC) {
    long long i = (long long)blockIdx.x * blockDim.x + threadIdx.x;
    if (i >= (long long)Nn * HC) return;
    int c = (int)(i % HC);
    float inv_n = 1.f / (float)Nn;
    float mu  = sums[c] * inv_n;
    float var = sqs[c] * inv_n - mu * mu;
    float y = (x[i] - mu) * rsqrtf(var + 1e-5f) * g[c] + be[c];
    x[i] = y > 0.f ? y : (__expf(y) - 1.f);
}

// ---------------------------------------------------------------------------
// Classifier: relu(x@Wc1 + bc1) @ Wc2 + bc2 ; weights cached in LDS
// ---------------------------------------------------------------------------
__global__ void k_classifier(const float* __restrict__ x,
                             const float* __restrict__ Wc1,
                             const float* __restrict__ bc1,
                             const float* __restrict__ Wc2,
                             const float* __restrict__ bc2,
                             float* __restrict__ out, int Nn) {
    __shared__ float w1[1024];
    __shared__ float b1[32];
    __shared__ float w2[32];
    int t = threadIdx.x;
    for (int i = t; i < 1024; i += blockDim.x) w1[i] = Wc1[i];
    if (t < 32) { b1[t] = bc1[t]; w2[t] = Wc2[t]; }
    __syncthreads();
    int n = blockIdx.x * blockDim.x + t;
    if (n >= Nn) return;
    const float* xr = x + (size_t)n * 32;
    float xi[32];
#pragma unroll
    for (int i = 0; i < 32; ++i) xi[i] = xr[i];
    float acc2 = bc2[0];
    for (int j = 0; j < 32; ++j) {
        float a = b1[j];
#pragma unroll
        for (int i = 0; i < 32; ++i) a += xi[i] * w1[i * 32 + j];
        a = a > 0.f ? a : 0.f;
        acc2 += a * w2[j];
    }
    out[n] = acc2;
}

// ---------------------------------------------------------------------------
// Host side
// ---------------------------------------------------------------------------
static inline unsigned cdiv_ll(long long a, long long b) { return (unsigned)((a + b - 1) / b); }

extern "C" void kernel_launch(void* const* d_in, const int* in_sizes, int n_in,
                              void* d_out, int out_size, void* d_ws, size_t ws_size,
                              hipStream_t stream) {
    const float* x   = (const float*)d_in[0];
    const int*   ei  = (const int*)  d_in[1];
    const float* W1  = (const float*)d_in[2];
    const float* as1 = (const float*)d_in[3];
    const float* ad1 = (const float*)d_in[4];
    const float* g1  = (const float*)d_in[6];
    const float* be1 = (const float*)d_in[7];
    const float* W2  = (const float*)d_in[8];
    const float* as2 = (const float*)d_in[9];
    const float* ad2 = (const float*)d_in[10];
    const float* g2  = (const float*)d_in[12];
    const float* be2 = (const float*)d_in[13];
    const float* W3  = (const float*)d_in[14];
    const float* as3 = (const float*)d_in[15];
    const float* ad3 = (const float*)d_in[16];
    const float* g3  = (const float*)d_in[18];
    const float* be3 = (const float*)d_in[19];
    const float* Wc1 = (const float*)d_in[20];
    const float* bc1 = (const float*)d_in[21];
    const float* Wc2 = (const float*)d_in[22];
    const float* bc2 = (const float*)d_in[23];

    const int IN = 128, H = 4, C = 64, HC = H * C;       // 256
    const int Nn = in_sizes[0] / IN;                     // 50000
    const long long E = (long long)in_sizes[1] / 2;      // 800000

    // workspace layout (floats)
    float* ws   = (float*)d_ws;
    float* bufA = ws;                                    // N*256
    float* bufB = bufA + (size_t)Nn * HC;                // N*256
    float* hbuf = bufB + (size_t)Nn * HC;                // N*256
    float* als  = hbuf + (size_t)Nn * HC;                // N*4
    float* ald  = als  + (size_t)Nn * H;
    float* mbuf = ald  + (size_t)Nn * H;
    float* dbuf = mbuf + (size_t)Nn * H;
    float* sums = dbuf + (size_t)Nn * H;                 // 256
    float* sqs  = sums + 256;                            // 256

    const float NEG_INF = -__builtin_inff();

    auto gat_layer = [&](const float* A, int K, const float* W,
                         const float* a_s, const float* a_d,
                         const float* g, const float* be,
                         int Hh, int Cc, float* outbuf) {
        int HCc = Hh * Cc;
        int mtile = (Nn + 15) / 16;
        unsigned mblocks = (unsigned)((mtile + 7) / 8);
        // h = A @ W   (8 waves / block, shared B panel in LDS)
        k_gemm_wmma<<<mblocks * (unsigned)(HCc / 16), 256, 0, stream>>>(A, W, hbuf, Nn, HCc, K);
        // attention logits
        k_logits<<<cdiv_ll((long long)Nn * Hh, 256), 256, 0, stream>>>(
            hbuf, a_s, a_d, als, ald, Nn, Hh, Cc);
        // init reducers / output
        k_fill<<<cdiv_ll((long long)Nn * Hh, 256), 256, 0, stream>>>(mbuf, (long long)Nn * Hh, NEG_INF);
        k_fill<<<cdiv_ll((long long)Nn * Hh, 256), 256, 0, stream>>>(dbuf, (long long)Nn * Hh, 0.f);
        k_fill<<<cdiv_ll((long long)Nn * HCc, 256), 256, 0, stream>>>(outbuf, (long long)Nn * HCc, 0.f);
        k_fill<<<cdiv_ll(512, 256), 256, 0, stream>>>(sums, 512, 0.f);   // sums + sqs
        // segment softmax
        long long totEH = (E + Nn) * Hh;
        k_edge_max<<<cdiv_ll(totEH, 256), 256, 0, stream>>>(ei, E, Nn, als, ald, mbuf, Hh);
        k_edge_denom<<<cdiv_ll(totEH, 256), 256, 0, stream>>>(ei, E, Nn, als, ald, mbuf, dbuf, Hh);
        // weighted message scatter-add (wave per edge)
        long long Etot = E + Nn;
        k_edge_scatter<<<cdiv_ll(Etot, 8), 256, 0, stream>>>(
            ei, E, Nn, als, ald, mbuf, dbuf, hbuf, outbuf, Hh, HCc);
        // batch norm + ELU (GAT bias cancels under BN)
        k_bn_stats<<<1024, 256, 0, stream>>>(outbuf, sums, sqs, Nn, HCc);
        k_bn_apply<<<cdiv_ll((long long)Nn * HCc, 256), 256, 0, stream>>>(
            outbuf, sums, sqs, g, be, Nn, HCc);
    };

    gat_layer(x,    IN, W1, as1, ad1, g1, be1, H, C,  bufA);   // 128 -> 4x64
    gat_layer(bufA, HC, W2, as2, ad2, g2, be2, H, C,  bufB);   // 256 -> 4x64
    gat_layer(bufB, HC, W3, as3, ad3, g3, be3, 1, 32, bufA);   // 256 -> 1x32

    k_classifier<<<cdiv_ll(Nn, 256), 256, 0, stream>>>(
        bufA, Wc1, bc1, Wc2, bc2, (float*)d_out, Nn);
}
